// BatchedLIDIA_86311662781105
// MI455X (gfx1250) — compile-verified, matrix operand order checked by reading, not run on hardware
//
#include <hip/hip_runtime.h>
#include <hip/hip_bf16.h>

// ---------------------------------------------------------------------------
// LIDIA / non-local-means denoiser for MI455X (gfx1250, wave32, WMMA).
//
// Shapes: t=4, c=3, H=W=128, PS=5 (D=75), WS=29 (841 offsets), K=14.
//
// Distance stage as f16 WMMA Gram tiles with f32 accumulation:
//   d(q,n) = |q|^2 + |n|^2 - 2<q,n>,  <q,n> via v_wmma_f32_16x16x32_f16.
//
// Feature K-dim is re-indexed as k = dy*16 + (dx*3+ch)  (K=80 pad 96) so the
// LDS offset of feature k is (k>>4)*156 + (k&15): pure shift/and, no table,
// no per-element branch.  Dummy K slots are zeroed in the A fragments only;
// B reads of dummy slots hit real-but-ignored f16 data (guaranteed finite --
// a zeroed LDS pad row -- since WMMA propagates NaN even against A==0).
// ---------------------------------------------------------------------------

typedef __attribute__((ext_vector_type(16))) _Float16 v16h;
typedef __attribute__((ext_vector_type(8)))  float    v8f;

#define T_   4
#define C_   3
#define H_   128
#define W_   128
#define PS_  5
#define D_   75          // PS*PS*C real features
#define WS_  29
#define WHALF_ 14
#define PHALF_ 2
#define PAD_ 16          // whalf + phalf
#define XP_  160         // H + 2*PAD
#define G_   156         // H + 2*WHALF (search grid extent)
#define KNN_ 14
#define ROWH_ 156        // halfs per LDS image row (52 cols * 3 ch)

// ---- workspace layout (bytes) ---------------------------------------------
#define WS_MEANS   0u                    // 12 f32
#define WS_XP      64u                   // 4*160*160*3 f32 = 1,228,800 B
#define WS_NNORM   1228864u              // 4*156*156 f32   =   389,376 B
#define WS_DK      1618240u              // 4*128*128*14 f32 = 3,670,016 B
#define WS_IDX     5288256u              // 4*128*128*14 i32 = 3,670,016 B
#define WS_PDENO   8958272u              // 4*128*128*75 f32 = 19,660,800 B
// total: 28,619,072 B (~27.3 MB)

// ---------------------------------------------------------------------------
// Kernel 1: per-(t,c) mean of normalized image.  x = v/127.5 - 1.
// ---------------------------------------------------------------------------
__global__ __launch_bounds__(256) void lidia_means(const float* __restrict__ noisy,
                                                   float* __restrict__ means) {
  __shared__ float red[256];
  const int b = blockIdx.x;                 // t*3 + c
  const float* p = noisy + (size_t)b * (H_ * W_);
  float s = 0.f;
  for (int i = threadIdx.x; i < H_ * W_; i += 256) s += p[i];
  red[threadIdx.x] = s;
  __syncthreads();
  for (int st = 128; st > 0; st >>= 1) {
    if (threadIdx.x < st) red[threadIdx.x] += red[threadIdx.x + st];
    __syncthreads();
  }
  if (threadIdx.x == 0)
    means[b] = red[0] * (1.0f / (H_ * W_)) * (1.0f / 127.5f) - 1.0f;
}

// ---------------------------------------------------------------------------
// Kernel 2: reflect-pad + normalize + de-mean -> xp[t][160][160][3] (f32).
// ---------------------------------------------------------------------------
__global__ __launch_bounds__(256) void lidia_pad(const float* __restrict__ noisy,
                                                 const float* __restrict__ means,
                                                 float* __restrict__ xp) {
  int i = blockIdx.x * blockDim.x + threadIdx.x;
  if (i >= T_ * XP_ * XP_) return;
  const int xx = i % XP_;
  const int yy = (i / XP_) % XP_;
  const int t  = i / (XP_ * XP_);
  int ry = yy - PAD_; ry = ry < 0 ? -ry : (ry > H_ - 1 ? 2 * (H_ - 1) - ry : ry);
  int rx = xx - PAD_; rx = rx < 0 ? -rx : (rx > W_ - 1 ? 2 * (W_ - 1) - rx : rx);
  #pragma unroll
  for (int ch = 0; ch < C_; ++ch) {
    float v = noisy[(((size_t)t * C_ + ch) * H_ + ry) * W_ + rx] * (1.0f / 127.5f)
              - 1.0f - means[t * C_ + ch];
    xp[(size_t)i * C_ + ch] = v;
  }
}

// ---------------------------------------------------------------------------
// Kernel 3: patch squared norms over the full search grid.
// ---------------------------------------------------------------------------
__global__ __launch_bounds__(256) void lidia_pnorm(const float* __restrict__ xp,
                                                   float* __restrict__ nn) {
  int i = blockIdx.x * blockDim.x + threadIdx.x;
  if (i >= T_ * G_ * G_) return;
  const int gx = i % G_;
  const int gy = (i / G_) % G_;
  const int t  = i / (G_ * G_);
  float s = 0.f;
  #pragma unroll
  for (int dy = 0; dy < PS_; ++dy)
    #pragma unroll
    for (int dx = 0; dx < PS_; ++dx)
      #pragma unroll
      for (int ch = 0; ch < C_; ++ch) {
        float v = xp[(((size_t)t * XP_ + gy + dy) * XP_ + gx + dx) * C_ + ch];
        s += v * v;
      }
  nn[i] = s;
}

// ---------------------------------------------------------------------------
// Kernel 4 (hot): WMMA distance field + running top-14.  One wave32 / block.
//   A (16x96 f16): query patches.   B (96x48 f16): neighbor columns.
//   G = A*B via 3(K)x3(N) v_wmma_f32_16x16x32_f16 per vertical offset oy.
// ---------------------------------------------------------------------------
__global__ __launch_bounds__(32) void lidia_dist_topk(const float* __restrict__ xp,
                                                      const float* __restrict__ nnorm,
                                                      float* __restrict__ dk_out,
                                                      int*   __restrict__ idx_out) {
  // 33 real rows + 1 zero pad row + 16 zero tail (dummy-K reads stay finite)
  __shared__ _Float16 s_xp16[34 * ROWH_ + 16];
  __shared__ float    s_g[16 * 48];
  __shared__ float    s_nn[48];

  const int lane = threadIdx.x;         // 0..31
  const int x0 = blockIdx.x * 16;
  const int y  = blockIdx.y;
  const int t  = blockIdx.z;

  // stage + convert the image strip (rows y..y+32, cols x0..x0+51, 3ch) to f16
  {
    const float* base = xp + (((size_t)t * XP_ + y) * XP_ + x0) * C_;
    for (int i = lane; i < 33 * ROWH_; i += 32) {
      int r = i / ROWH_, c = i % ROWH_;
      s_xp16[i] = (_Float16)base[(size_t)r * (XP_ * C_) + c];
    }
    for (int i = lane; i < ROWH_ + 16; i += 32)
      s_xp16[33 * ROWH_ + i] = (_Float16)0.f;
  }
  __syncthreads();

  const int n_local = lane & 15;
  const int hi8     = (lane >> 4) * 8;  // K offset for upper half-wave

  // ---- A fragments (query patches), built once -----------------------------
  // k-pair base: kb = kt*32 + (j>>2)*16 + (j&3)*2 + hi8 ; lds off = (kb>>4)*156+(kb&15)
  v16h afrag[3];
  {
    const int abase = WHALF_ * ROWH_ + (WHALF_ + n_local) * 3 + hi8;
    #pragma unroll
    for (int kt = 0; kt < 3; ++kt) {
      #pragma unroll
      for (int j = 0; j < 8; ++j) {
        const int koff = (kt * 2 + (j >> 2)) * ROWH_ + (j & 3) * 2;  // compile-time
        const int kc   = kt * 32 + (j >> 2) * 16 + (j & 3) * 2;      // + hi8 runtime
        _Float16 v0 = s_xp16[abase + koff];
        _Float16 v1 = s_xp16[abase + koff + 1];
        const int k0 = kc + hi8;
        const int k1 = k0 + 1;
        const bool r0 = (k0 < 80);                         // even k: (k&15)<15 always
        const bool r1 = (k1 < 80) && ((k1 & 15) != 15);    // w==15 slots are dummy
        afrag[kt][2 * j]     = r0 ? v0 : (_Float16)0.f;
        afrag[kt][2 * j + 1] = r1 ? v1 : (_Float16)0.f;
      }
    }
  }

  // query patch norm for this lane's query pixel
  const float qn = nnorm[((size_t)t * G_ + (y + WHALF_)) * G_ + (x0 + WHALF_ + n_local)];

  // ---- running top-K (registers, fully unrolled predicated updates) --------
  float bd[KNN_];
  int   bi[KNN_];
  #pragma unroll
  for (int k = 0; k < KNN_; ++k) { bd[k] = 3.4e38f; bi[k] = 0; }
  float curMax = 3.4e38f;
  int   maxPos = 0;

  const v8f vzero = {0.f, 0.f, 0.f, 0.f, 0.f, 0.f, 0.f, 0.f};

  for (int oy = 0; oy < WS_; ++oy) {
    for (int i = lane; i < 48; i += 32)
      s_nn[i] = nnorm[((size_t)t * G_ + (y + oy)) * G_ + (x0 + i)];
    if (oy + 1 < WS_)
      __builtin_prefetch(&nnorm[((size_t)t * G_ + (y + oy + 1)) * G_ + x0], 0, 0);
    __syncthreads();

    v8f acc[3] = {vzero, vzero, vzero};

    #pragma unroll
    for (int kt = 0; kt < 3; ++kt) {          // kt outer: consecutive WMMAs hit
      #pragma unroll
      for (int nt = 0; nt < 3; ++nt) {        // different accumulators
        const int bbase = oy * ROWH_ + (nt * 16 + n_local) * 3 + hi8;
        v16h b;
        #pragma unroll
        for (int j = 0; j < 8; ++j) {
          const int koff = (kt * 2 + (j >> 2)) * ROWH_ + (j & 3) * 2;  // imm offset
          b[2 * j]     = s_xp16[bbase + koff];        // ds_load_u16 (d16)
          b[2 * j + 1] = s_xp16[bbase + koff + 1];
        }
        acc[nt] = __builtin_amdgcn_wmma_f32_16x16x32_f16(
            /*neg_a=*/false, afrag[kt], /*neg_b=*/false, b,
            /*c_mod=*/(short)0, acc[nt], /*reuse_a=*/false, /*reuse_b=*/false);
      }
    }

    // spill Gram tile: lane holds (M = r + hi8, N = nt*16 + n_local)
    #pragma unroll
    for (int nt = 0; nt < 3; ++nt) {
      const int N = nt * 16 + n_local;
      #pragma unroll
      for (int r = 0; r < 8; ++r)
        s_g[(r + hi8) * 48 + N] = acc[nt][r];
    }
    __syncthreads();

    // banded extraction + top-K (one query per lane, lanes 0..15)
    if (lane < 16) {
      const int q = lane;
      for (int ox = 0; ox < WS_; ++ox) {
        float g = s_g[q * 48 + q + ox];
        float d = qn + s_nn[q + ox] - 2.0f * g;
        int   o = oy * WS_ + ox;
        if (d < curMax) {
          #pragma unroll
          for (int kk = 0; kk < KNN_; ++kk)
            if (kk == maxPos) { bd[kk] = d; bi[kk] = o; }
          curMax = bd[0]; maxPos = 0;
          #pragma unroll
          for (int kk = 1; kk < KNN_; ++kk)
            if (bd[kk] > curMax) { curMax = bd[kk]; maxPos = kk; }
        }
      }
    }
    __syncthreads();  // protect s_nn/s_g before next iteration
  }

  if (lane < 16) {
    const size_t base = (((size_t)t * H_ + y) * W_ + (x0 + lane)) * KNN_;
    #pragma unroll
    for (int kk = 0; kk < KNN_; ++kk) {
      dk_out[base + kk]  = bd[kk];
      idx_out[base + kk] = bi[kk];
    }
  }
}

// ---------------------------------------------------------------------------
// Kernel 5: softmax weights over K + weighted neighbor-patch aggregation.
// One block per pixel; lanes 0..74 each own one patch feature.
// ---------------------------------------------------------------------------
__global__ __launch_bounds__(96) void lidia_agg(const float* __restrict__ xp,
                                                const float* __restrict__ dk,
                                                const int*   __restrict__ idx,
                                                const int*   __restrict__ sigma_p,
                                                float* __restrict__ pdeno) {
  __shared__ float s_w[KNN_];
  __shared__ int   s_o[KNN_];
  const int pid = blockIdx.x;                  // t*H*W + y*W + x
  const int x = pid & (W_ - 1);
  const int y = (pid >> 7) & (H_ - 1);
  const int t = pid >> 14;
  const size_t kb = (size_t)pid * KNN_;

  if (threadIdx.x < KNN_) s_o[threadIdx.x] = idx[kb + threadIdx.x];
  if (threadIdx.x == 0) {
    const float sig  = (float)(*sigma_p) * (1.0f / 255.0f) * 2.0f;  // /255/0.5
    const float beta = 1.0f / (2.0f * sig * sig * (float)D_);
    float dv[KNN_];
    float dmin = 3.4e38f;
    #pragma unroll
    for (int k = 0; k < KNN_; ++k) { dv[k] = dk[kb + k]; dmin = fminf(dmin, dv[k]); }
    float s = 0.f;
    #pragma unroll
    for (int k = 0; k < KNN_; ++k) { float e = __expf(-beta * (dv[k] - dmin));
                                     s_w[k] = e; s += e; }
    const float inv = 1.0f / s;
    #pragma unroll
    for (int k = 0; k < KNN_; ++k) s_w[k] *= inv;
  }
  __syncthreads();

  const int d = threadIdx.x;
  if (d < D_) {
    const int dy = d / 15, rm = d % 15, dx = rm / 3, ch = rm % 3;
    float a = 0.f;
    #pragma unroll
    for (int k = 0; k < KNN_; ++k) {
      const int o  = s_o[k];
      const int oy = o / WS_, ox = o % WS_;
      a += s_w[k] * xp[(((size_t)t * XP_ + (y + oy + dy)) * XP_ + (x + ox + dx)) * C_ + ch];
    }
    pdeno[(size_t)pid * D_ + d] = a;
  }
}

// ---------------------------------------------------------------------------
// Kernel 6: overlap-add fold + un-normalize.  Output layout [t][c][H][W] f32.
// ---------------------------------------------------------------------------
__global__ __launch_bounds__(256) void lidia_fold(const float* __restrict__ pdeno,
                                                  const float* __restrict__ means,
                                                  float* __restrict__ out) {
  int i = blockIdx.x * blockDim.x + threadIdx.x;
  if (i >= T_ * C_ * H_ * W_) return;
  const int X  = i & (W_ - 1);
  const int Y  = (i >> 7) & (H_ - 1);
  const int ch = (i >> 14) % C_;
  const int t  = i / (C_ * H_ * W_);
  float a = 0.f;
  int cnt = 0;
  #pragma unroll
  for (int py = 0; py < PS_; ++py) {
    const int h = Y + PHALF_ - py;
    if ((unsigned)h >= (unsigned)H_) continue;
    #pragma unroll
    for (int px = 0; px < PS_; ++px) {
      const int w = X + PHALF_ - px;
      if ((unsigned)w >= (unsigned)W_) continue;
      a += pdeno[(((size_t)t * H_ + h) * W_ + w) * D_ + (py * PS_ + px) * C_ + ch];
      ++cnt;
    }
  }
  const float deno = a / (float)cnt + means[t * C_ + ch];
  out[i] = 255.0f * (deno * 0.5f + 0.5f);
}

// ---------------------------------------------------------------------------
extern "C" void kernel_launch(void* const* d_in, const int* in_sizes, int n_in,
                              void* d_out, int out_size, void* d_ws, size_t ws_size,
                              hipStream_t stream) {
  const float* noisy   = (const float*)d_in[0];   // [4,3,128,128] f32
  const int*   sigma_p = (const int*)d_in[1];     // scalar int

  char* ws = (char*)d_ws;                         // needs ~27.3 MB
  float* means = (float*)(ws + WS_MEANS);
  float* xp    = (float*)(ws + WS_XP);
  float* nn    = (float*)(ws + WS_NNORM);
  float* dk    = (float*)(ws + WS_DK);
  int*   idx   = (int*)  (ws + WS_IDX);
  float* pdeno = (float*)(ws + WS_PDENO);
  float* out   = (float*)d_out;

  lidia_means<<<T_ * C_, 256, 0, stream>>>(noisy, means);
  lidia_pad<<<(T_ * XP_ * XP_ + 255) / 256, 256, 0, stream>>>(noisy, means, xp);
  lidia_pnorm<<<(T_ * G_ * G_ + 255) / 256, 256, 0, stream>>>(xp, nn);
  lidia_dist_topk<<<dim3(W_ / 16, H_, T_), 32, 0, stream>>>(xp, nn, dk, idx);
  lidia_agg<<<T_ * H_ * W_, 96, 0, stream>>>(xp, dk, idx, sigma_p, pdeno);
  lidia_fold<<<(T_ * C_ * H_ * W_ + 255) / 256, 256, 0, stream>>>(pdeno, means, out);
}
